// DetectPostProcess_13134009991469
// MI455X (gfx1250) — compile-verified, matrix operand order checked by reading, not run on hardware
//
#include <hip/hip_runtime.h>

// ---- problem constants (match reference) ----
#define B_     32
#define A_     32768
#define NCP1   21
#define NC_    20
#define KTOP   200
#define KPAD   256
#define VARC   0.125f
#define THCONF 0.05f
#define THIOU  0.5f

typedef __attribute__((ext_vector_type(16))) _Float16 v16h;
typedef __attribute__((ext_vector_type(8)))  float    v8f;

// ============================================================
// Phase 1: one coalesced pass over conf/loc. Computes per-anchor
// softmax max + RECIPROCAL denominator and decoded xyxy boxes.
// ~104 MB streamed from HBM -> bandwidth bound (~4.5us at 23.3 TB/s).
// ============================================================
__global__ __launch_bounds__(256) void decode_stats_kernel(
    const float* __restrict__ conf, const float* __restrict__ loc,
    const float* __restrict__ anchor, float* __restrict__ box,
    float* __restrict__ maxv, float* __restrict__ rdenom)
{
    int idx = blockIdx.x * blockDim.x + threadIdx.x;
    if (idx >= B_ * A_) return;
    int a = idx & (A_ - 1);

    size_t cbase = (size_t)idx * NCP1;
    // CDNA5 prefetch path (global_prefetch_b8): pull upcoming conf rows
    __builtin_prefetch(conf + cbase + (size_t)NCP1 * 4096, 0, 1);

    float m = conf[cbase];
    #pragma unroll
    for (int c = 1; c < NCP1; ++c) m = fmaxf(m, conf[cbase + c]);
    float s = 0.0f;
    #pragma unroll
    for (int c = 0; c < NCP1; ++c) s += expf(conf[cbase + c] - m);
    maxv[idx]   = m;
    rdenom[idx] = 1.0f / s;   // one divide per anchor; phase 2 multiplies

    const float* lp = loc    + (size_t)idx * 4;
    const float* ap = anchor + (size_t)a   * 4;
    float l0 = lp[0], l1 = lp[1], l2 = lp[2], l3 = lp[3];
    float acx = ap[0], acy = ap[1], aw = ap[2], ah = ap[3];
    float cx = l0 * VARC * aw + acx;
    float cy = l1 * VARC * ah + acy;
    float w  = expf(l2 * VARC) * aw;
    float h  = expf(l3 * VARC) * ah;
    float* bp = box + (size_t)idx * 4;
    bp[0] = cx - 0.5f * w;
    bp[1] = cy - 0.5f * h;
    bp[2] = cx + 0.5f * w;
    bp[3] = cy + 0.5f * h;
}

// ============================================================
// Phase 2: per (image,class) pair (640 blocks) running bitonic
// top-256 selection over 32768 thresholded softmax scores.
// conf re-reads are class-strided but all of conf (88MB) fits in
// the 192MB L2, so this phase is L2-resident.
// ============================================================
__global__ __launch_bounds__(256) void topk_kernel(
    const float* __restrict__ conf, const float* __restrict__ maxv,
    const float* __restrict__ rdenom, float* __restrict__ topv,
    int* __restrict__ topi)
{
    __shared__ float sv[KPAD]; __shared__ int si[KPAD];
    __shared__ float cv[KPAD]; __shared__ int ci[KPAD];

    const int pair = blockIdx.x;        // b*NC + c
    const int b    = pair / NC_;
    const int cls  = pair - b * NC_;    // 0..19 -> conf channel cls+1
    const int t    = threadIdx.x;

    sv[t] = -2.0f;  // below masked value (-1) so any real element wins
    si[t] = 0;
    __syncthreads();

    for (int chunk = 0; chunk < A_ / KPAD; ++chunk) {
        int a = chunk * KPAD + t;
        size_t ai = (size_t)b * A_ + a;
        float x  = conf[ai * NCP1 + cls + 1];
        float sc = expf(x - maxv[ai]) * rdenom[ai];
        cv[t] = (sc >= THCONF) ? sc : -1.0f;   // masked like reference
        ci[t] = a;
        __syncthreads();

        // bitonic sort chunk, descending
        for (int k = 2; k <= KPAD; k <<= 1) {
            for (int j = k >> 1; j > 0; j >>= 1) {
                int p = t ^ j;
                if (p > t) {
                    bool desc = ((t & k) == 0);
                    float av = cv[t], bv = cv[p];
                    if (desc ? (av < bv) : (av > bv)) {
                        cv[t] = bv; cv[p] = av;
                        int tmp = ci[t]; ci[t] = ci[p]; ci[p] = tmp;
                    }
                }
                __syncthreads();
            }
        }

        // merge: max(L[t], M[255-t]) = 256 largest, bitonic sequence
        float mv = cv[KPAD - 1 - t]; int mi = ci[KPAD - 1 - t];
        if (mv > sv[t]) { sv[t] = mv; si[t] = mi; }
        __syncthreads();
        // clean bitonic sequence -> descending
        for (int j = KPAD >> 1; j > 0; j >>= 1) {
            int p = t ^ j;
            if (p > t) {
                float av = sv[t], bv = sv[p];
                if (av < bv) {
                    sv[t] = bv; sv[p] = av;
                    int tmp = si[t]; si[t] = si[p]; si[p] = tmp;
                }
            }
            __syncthreads();
        }
    }

    topv[(size_t)pair * KPAD + t] = sv[t];
    topi[(size_t)pair * KPAD + t] = si[t];
}

// ============================================================
// Phase 3: NMS per (image,class). The IoU denominator's
// area_i + area_j term is a rank-2 outer product:
//   [area_i, 1] @ [[1...],[area_j...]]
// computed on the WMMA pipe (v_wmma_f32_16x16x32_f16), one 16x16
// tile per wave step over the padded 256x256 candidate matrix.
// The IoU threshold is evaluated division-free:
//   inter/(union+eps) > TH  <=>  inter > TH*(union+eps)   (union+eps > 0)
// Suppression rows stored as LDS bitmasks; serial scan matches
// the reference lax.scan semantics.
// ============================================================
__global__ __launch_bounds__(256) void nms_kernel(
    const float* __restrict__ box, const float* __restrict__ topv,
    const int* __restrict__ topi, float* __restrict__ out)
{
    __shared__ float    bxs[KPAD * 4];
    __shared__ float    sarea[KPAD];
    __shared__ float    tsv[KPAD];
    __shared__ unsigned rows[KPAD * 8];   // 256-bit row masks: iou > TH
    __shared__ unsigned char keepf[KPAD];

    const int pair = blockIdx.x;
    const int b    = pair / NC_;
    const int t    = threadIdx.x;

    #pragma unroll
    for (int w = 0; w < 8; ++w) rows[t * 8 + w] = 0u;

    if (t < KTOP) {
        int idx = topi[(size_t)pair * KPAD + t];
        const float* bp = box + ((size_t)b * A_ + idx) * 4;
        bxs[t * 4 + 0] = bp[0]; bxs[t * 4 + 1] = bp[1];
        bxs[t * 4 + 2] = bp[2]; bxs[t * 4 + 3] = bp[3];
        tsv[t] = topv[(size_t)pair * KPAD + t];
    } else {
        bxs[t * 4 + 0] = 0.f; bxs[t * 4 + 1] = 0.f;
        bxs[t * 4 + 2] = 0.f; bxs[t * 4 + 3] = 0.f;
        tsv[t] = -1.0f;
    }
    sarea[t] = (bxs[t * 4 + 2] - bxs[t * 4 + 0]) *
               (bxs[t * 4 + 3] - bxs[t * 4 + 1]);
    keepf[t] = 0;
    __syncthreads();

    const int wave = t >> 5;      // 8 wave32 waves
    const int lane = t & 31;

    // 16x16 tile grid over padded 256x256 -> 256 tiles, 32 per wave
    for (int tile = wave; tile < 256; tile += 8) {
        int ti = tile >> 4, tj = tile & 15;
        int i0 = ti * 16, j0 = tj * 16;

        // A (16x32 f16): row m = [area_m, 1, 0, ...]
        // lane<16 holds M=lane, K=0..7 in h0..h7 -> h0,h1 used
        v16h af = {};
        if (lane < 16) {
            af[0] = (_Float16)sarea[i0 + lane];
            af[1] = (_Float16)1.0f;
        }
        // B (32x16 f16): col n = [1, area_n, 0, ...]^T
        // lane<16 holds N=lane, K=0..7 in h0..h7 -> h0=K0, h1=K1 used
        v16h bf = {};
        if (lane < 16) {
            bf[0] = (_Float16)1.0f;
            bf[1] = (_Float16)sarea[j0 + lane];
        }
        v8f c = {};
        c = __builtin_amdgcn_wmma_f32_16x16x32_f16(
                false, af, false, bf, (short)0, c, false, false);
        // D[m][n] = area_{i0+m} + area_{j0+n}

        int nloc  = lane & 15;
        int mbase = (lane < 16) ? 0 : 8;
        #pragma unroll
        for (int r = 0; r < 8; ++r) {
            int gi = i0 + mbase + r;
            int gj = j0 + nloc;
            float asum = c[r];
            float lx = fmaxf(bxs[gi * 4 + 0], bxs[gj * 4 + 0]);
            float ly = fmaxf(bxs[gi * 4 + 1], bxs[gj * 4 + 1]);
            float rx = fminf(bxs[gi * 4 + 2], bxs[gj * 4 + 2]);
            float ry = fminf(bxs[gi * 4 + 3], bxs[gj * 4 + 3]);
            float iw = fmaxf(rx - lx, 0.0f);
            float ih = fmaxf(ry - ly, 0.0f);
            float inter = iw * ih;
            // division-free: inter/(asum-inter+eps) > TH
            if (inter > THIOU * (asum - inter + 1e-9f))
                atomicOr(&rows[gi * 8 + (gj >> 5)], 1u << (gj & 31));
        }
    }
    __syncthreads();

    // sequential suppression scan (reference lax.scan semantics)
    if (t == 0) {
        unsigned supp[8] = {0u, 0u, 0u, 0u, 0u, 0u, 0u, 0u};
        for (int i = 0; i < KTOP; ++i) {
            bool k = (tsv[i] >= THCONF) &&
                     !((supp[i >> 5] >> (i & 31)) & 1u);
            keepf[i] = k ? 1 : 0;
            if (k) {
                #pragma unroll
                for (int w = 0; w < 8; ++w) supp[w] |= rows[i * 8 + w];
            }
        }
    }
    __syncthreads();

    if (t < KTOP) {
        bool k = keepf[t] != 0;
        size_t ob = ((size_t)pair * KTOP + t) * 5;
        out[ob + 0] = k ? bxs[t * 4 + 0] : 0.0f;
        out[ob + 1] = k ? bxs[t * 4 + 1] : 0.0f;
        out[ob + 2] = k ? bxs[t * 4 + 2] : 0.0f;
        out[ob + 3] = k ? bxs[t * 4 + 3] : 0.0f;
        out[ob + 4] = k ? tsv[t] : 0.0f;
        // keep[] flat after objs (B*NC*K*5 = 640000 floats)
        out[(size_t)B_ * NC_ * KTOP * 5 + (size_t)pair * KTOP + t] =
            k ? 1.0f : 0.0f;
    }
}

extern "C" void kernel_launch(void* const* d_in, const int* in_sizes, int n_in,
                              void* d_out, int out_size, void* d_ws, size_t ws_size,
                              hipStream_t stream) {
    const float* conf   = (const float*)d_in[0];  // [32, 32768, 21]
    const float* loc    = (const float*)d_in[1];  // [32, 32768, 4]
    const float* anchor = (const float*)d_in[2];  // [32768, 4]
    float* out = (float*)d_out;                   // 640000 objs + 128000 keep

    // workspace layout (floats): box | maxv | rdenom | topv | topi
    float* wsf    = (float*)d_ws;
    float* box    = wsf;                                    // B*A*4
    float* maxv   = box    + (size_t)B_ * A_ * 4;           // B*A
    float* rdenom = maxv   + (size_t)B_ * A_;               // B*A
    float* topv   = rdenom + (size_t)B_ * A_;               // 640*256
    int*   topi   = (int*)(topv + (size_t)B_ * NC_ * KPAD); // 640*256

    dim3 blk(256);
    decode_stats_kernel<<<dim3((B_ * A_ + 255) / 256), blk, 0, stream>>>(
        conf, loc, anchor, box, maxv, rdenom);
    topk_kernel<<<dim3(B_ * NC_), blk, 0, stream>>>(
        conf, maxv, rdenom, topv, topi);
    nms_kernel<<<dim3(B_ * NC_), blk, 0, stream>>>(
        box, topv, topi, out);
}
